// GnnMlpActor_46729244180499
// MI455X (gfx1250) — matrix operand order repaired and checked by reading, not compiled, not run on hardware
//
#include <hip/hip_runtime.h>
#include <hip/hip_bf16.h>

#define N_NODES 20000
#define N_EDGES 320000
#define OBS 256
#define HID 256
#define EMB 128
#define MLP_H 512
#define ACT_D 64

typedef __attribute__((ext_vector_type(4)))  unsigned int v4u;
typedef __attribute__((ext_vector_type(8)))  float        v8f;
typedef __attribute__((ext_vector_type(16))) __bf16       v16bf;

union BFrag { v4u q[2]; v16bf v; };

__device__ __forceinline__ unsigned short f2bf(float f) {
  unsigned int u = __float_as_uint(f);
  u += 0x7FFFu + ((u >> 16) & 1u);            // round-to-nearest-even
  return (unsigned short)(u >> 16);
}

// ---------------- degree / normalization ----------------
__global__ void deg_init_kernel(float* a, float* b, int n) {
  int i = blockIdx.x * 256 + threadIdx.x;
  if (i < n) { a[i] = 1.0f; b[i] = 1.0f; }     // self-loop contributes 1 to both
}

__global__ void deg_edge_kernel(const int* __restrict__ src, const int* __restrict__ dst,
                                float* dO, float* dI, int e) {
  int i = blockIdx.x * 256 + threadIdx.x;
  if (i < e) { atomicAdd(&dO[src[i]], 1.0f); atomicAdd(&dI[dst[i]], 1.0f); }
}

__global__ void deg_rsqrt_kernel(float* a, float* b, int n) {
  int i = blockIdx.x * 256 + threadIdx.x;
  if (i < n) { a[i] = rsqrtf(fmaxf(a[i], 1.0f)); b[i] = rsqrtf(fmaxf(b[i], 1.0f)); }
}

// ---------------- elementwise: scale + cvt to bf16 ----------------
__global__ void scale_bf16_kernel(const float* __restrict__ X, const float* __restrict__ dinv,
                                  unsigned short* __restrict__ out, int total, int shift) {
  int i = blockIdx.x * 256 + threadIdx.x;
  if (i < total) out[i] = f2bf(X[i] * dinv[i >> shift]);
}

// ---------------- weight packing into WMMA B-fragment layout ----------------
//   Bpk[(kt*Nout + n)*32 + g*16 + e] = W[(kt*32 + g*16 + e)*Nout + n]
__global__ void packB_kernel(const float* __restrict__ W, unsigned short* __restrict__ Bpk,
                             int total, int Nout, int ns) {
  int i = blockIdx.x * 256 + threadIdx.x;
  if (i >= total) return;
  int k = i >> ns;
  int n = i & (Nout - 1);
  int kt = k >> 5, g = (k >> 4) & 1, e = k & 15;
  Bpk[((size_t)kt * Nout + n) * 32 + (g << 4) + e] = f2bf(W[i]);
}

// ---------------- WMMA GEMM: C[M,Nout] = A[M,K](bf16) * B(bf16) (+bias)(+relu) ----
// Block = 256 threads = 8 waves; block tile 128(M) x 64(N); wave tile 16 x 64.
__global__ void __launch_bounds__(256)
wmma_gemm(const unsigned short* __restrict__ A,
          const unsigned short* __restrict__ Bpk,
          const float* __restrict__ bias,
          float* __restrict__ outF,
          unsigned short* __restrict__ outB,
          int K, int Nout, int act) {
  __shared__ unsigned short sB[64 * 32];      // one 64-col x 32-K bf16 tile (4 KB)
  const int lane = threadIdx.x & 31;
  const int wave = threadIdx.x >> 5;
  const int g    = lane >> 4;                 // half-wave group
  const int l16  = lane & 15;
  const int row0 = blockIdx.y * 128 + wave * 16;
  const int col0 = blockIdx.x * 64;

  int arow = row0 + l16;
  if (arow >= N_NODES) arow = N_NODES - 1;    // clamp loads; stores are guarded
  const unsigned short* aptr = A + (size_t)arow * K + (g << 3);

  v8f acc0 = {}, acc1 = {}, acc2 = {}, acc3 = {};
  const int ktiles = K >> 5;
  for (int kt = 0; kt < ktiles; ++kt) {
    // cooperative B tile load: 256 threads x 16B = 4KB, layout-preserving
    const v4u* gBt = (const v4u*)(Bpk + ((size_t)kt * Nout + col0) * 32);
    ((v4u*)sB)[threadIdx.x] = gBt[threadIdx.x];

    // A fragment: ISA 16-bit A layout -> lanes 0-15: K = {0..7,16..23}+kt*32,
    // lanes 16-31: +8. Two 16B loads per lane.
    BFrag a;
    const v4u* ap = (const v4u*)(aptr + (kt << 5));
    a.q[0] = ap[0];
    a.q[1] = ap[2];
    if (kt + 1 < ktiles)                       // pull next A tile into WGP cache
      __builtin_prefetch((const void*)(aptr + ((kt + 1) << 5)), 0, 3);

    __syncthreads();
    // load ALL four B fragments first -> 8 ds_load_b128 in flight, one wait,
    // then four back-to-back WMMAs
    const v4u* sv = (const v4u*)sB;
    const int bofs = l16 * 4 + g;              // v4u units
    BFrag b0, b1, b2, b3;
    b0.q[0] = sv[bofs      ]; b0.q[1] = sv[bofs +   2];
    b1.q[0] = sv[bofs +  64]; b1.q[1] = sv[bofs +  66];
    b2.q[0] = sv[bofs + 128]; b2.q[1] = sv[bofs + 130];
    b3.q[0] = sv[bofs + 192]; b3.q[1] = sv[bofs + 194];
    acc0 = __builtin_amdgcn_wmma_f32_16x16x32_bf16(false, a.v, false, b0.v, (short)0, acc0, false, false);
    acc1 = __builtin_amdgcn_wmma_f32_16x16x32_bf16(false, a.v, false, b1.v, (short)0, acc1, false, false);
    acc2 = __builtin_amdgcn_wmma_f32_16x16x32_bf16(false, a.v, false, b2.v, (short)0, acc2, false, false);
    acc3 = __builtin_amdgcn_wmma_f32_16x16x32_bf16(false, a.v, false, b3.v, (short)0, acc3, false, false);
    __syncthreads();
  }

  float bsv[4] = {0.f, 0.f, 0.f, 0.f};
  if (bias) {
#pragma unroll
    for (int j = 0; j < 4; ++j) bsv[j] = bias[col0 + j * 16 + l16];
  }
  v8f acc[4] = {acc0, acc1, acc2, acc3};
#pragma unroll
  for (int j = 0; j < 4; ++j) {
    const int col = col0 + j * 16 + l16;
#pragma unroll
    for (int r = 0; r < 8; ++r) {
      const int row = row0 + (g << 3) + r;    // C/D layout: VGPR r -> M = r + g*8
      if (row < N_NODES) {
        float v = acc[j][r] + bsv[j];
        if (act == 1) v = fmaxf(v, 0.0f);
        const size_t o = (size_t)row * Nout + col;
        if (outF) outF[o] = v;
        if (outB) outB[o] = f2bf(v);
      }
    }
  }
}

// ---------------- edge scatter-add: m[dst] += hw[src], float4 per thread ----------
__global__ void scatter_kernel(const float* __restrict__ hw, float* __restrict__ m,
                               const int* __restrict__ src, const int* __restrict__ dst,
                               int total4, int shift, int qmask) {
  int i = blockIdx.x * 256 + threadIdx.x;      // one thread = 4 consecutive feats
  if (i >= total4) return;
  int e  = i >> (shift - 2);
  int f0 = (i & qmask) << 2;
  const float* s = hw + ((size_t)src[e] << shift) + f0;
  float*       d = m  + ((size_t)dst[e] << shift) + f0;
  float4 v = *(const float4*)s;
  atomicAdd(d + 0, v.x);
  atomicAdd(d + 1, v.y);
  atomicAdd(d + 2, v.z);
  atomicAdd(d + 3, v.w);
}

// ---------------- GCN epilogue: act(m*dinv_in + b), emit f32 and/or scaled bf16 ----
__global__ void gcn_finish_kernel(const float* __restrict__ m, const float* __restrict__ dinvI,
                                  const float* __restrict__ bias, const float* __restrict__ dinvO,
                                  float* __restrict__ outF, unsigned short* __restrict__ outB,
                                  int total, int shift, int mask, int act) {
  int i = blockIdx.x * 256 + threadIdx.x;
  if (i >= total) return;
  int row = i >> shift, f = i & mask;
  float v = fmaf(m[i], dinvI[row], bias[f]);
  if (act == 1) v = fmaxf(v, 0.0f);
  else          v = 1.0f / (1.0f + __expf(-v)) + 1e-8f;   // positive_safe_sigmoid
  if (outF) outF[i] = v;
  float s = dinvO ? v * dinvO[row] : v;
  outB[i] = f2bf(s);
}

// ---------------- softmax over 64 classes: one wave per row ----------------
__global__ void softmax64_kernel(const float* __restrict__ logits, float* __restrict__ prob,
                                 int nrows) {
  int lane = threadIdx.x & 31;
  int row  = blockIdx.x * 8 + (threadIdx.x >> 5);
  if (row >= nrows) return;
  const float* L = logits + (size_t)row * 64;
  float a = L[lane], b = L[lane + 32];
  float mx = fmaxf(a, b);
  for (int o = 16; o > 0; o >>= 1) mx = fmaxf(mx, __shfl_xor(mx, o, 32));
  float ea = __expf(a - mx), eb = __expf(b - mx);
  float s = ea + eb;
  for (int o = 16; o > 0; o >>= 1) s += __shfl_xor(s, o, 32);
  float inv = 1.0f / s;
  float* P = prob + (size_t)row * 64;
  P[lane] = ea * inv;
  P[lane + 32] = eb * inv;
}

// ---------------- host ----------------
extern "C" void kernel_launch(void* const* d_in, const int* in_sizes, int n_in,
                              void* d_out, int out_size, void* d_ws, size_t ws_size,
                              hipStream_t stream) {
  (void)in_sizes; (void)n_in; (void)out_size; (void)ws_size;
  const float* X   = (const float*)d_in[0];
  const int*   src = (const int*)d_in[1];
  const int*   dst = (const int*)d_in[2];
  const float* Wg1 = (const float*)d_in[3],  *bg1 = (const float*)d_in[4];
  const float* Wg2 = (const float*)d_in[5],  *bg2 = (const float*)d_in[6];
  const float* Wg3 = (const float*)d_in[7],  *bg3 = (const float*)d_in[8];
  const float* W1  = (const float*)d_in[9],  *b1  = (const float*)d_in[10];
  const float* W2  = (const float*)d_in[11], *b2  = (const float*)d_in[12];
  const float* W3  = (const float*)d_in[13], *b3  = (const float*)d_in[14];

  float* prob = (float*)d_out;
  float* gnn  = prob + (size_t)N_NODES * ACT_D;   // outputs: prob, then gnn_output

  char* ws = (char*)d_ws;
  size_t off = 0;
  auto alloc = [&](size_t bytes) -> void* {
    void* p = ws + off; off += (bytes + 255) & ~(size_t)255; return p;
  };
  float*          dinvO = (float*)alloc((size_t)N_NODES * 4);
  float*          dinvI = (float*)alloc((size_t)N_NODES * 4);
  unsigned short* actA  = (unsigned short*)alloc((size_t)N_NODES * 512 * 2);
  unsigned short* actB  = (unsigned short*)alloc((size_t)N_NODES * 512 * 2);
  float*          hw    = (float*)alloc((size_t)N_NODES * 512 * 4);
  float*          m     = (float*)alloc((size_t)N_NODES * 256 * 4);
  unsigned short* wpk   = (unsigned short*)alloc((size_t)512 * 512 * 2);

  auto nb = [](int total) { return (total + 255) / 256; };

  // degree normalization (recomputed every call; deterministic)
  deg_init_kernel <<<nb(N_NODES), 256, 0, stream>>>(dinvO, dinvI, N_NODES);
  deg_edge_kernel <<<nb(N_EDGES), 256, 0, stream>>>(src, dst, dinvO, dinvI, N_EDGES);
  deg_rsqrt_kernel<<<nb(N_NODES), 256, 0, stream>>>(dinvO, dinvI, N_NODES);

  // h0 scaled + converted to bf16
  scale_bf16_kernel<<<nb(N_NODES * OBS), 256, 0, stream>>>(X, dinvO, actA, N_NODES * OBS, 8);

  auto gemm = [&](const unsigned short* A, const float* W, int K, int Nout, int ns,
                  const float* bias, float* oF, unsigned short* oB, int act) {
    packB_kernel<<<nb(K * Nout), 256, 0, stream>>>(W, wpk, K * Nout, Nout, ns);
    dim3 grid(Nout / 64, (N_NODES + 127) / 128);
    wmma_gemm<<<grid, 256, 0, stream>>>(A, wpk, bias, oF, oB, K, Nout, act);
  };
  auto gcn_rest = [&](int dim, int shift, const float* bias, const float* scaleNext,
                      float* oF, unsigned short* oB, int act) {
    // self-loop term: m = hw, then scatter original edges
    hipMemcpyAsync(m, hw, (size_t)N_NODES * dim * 4, hipMemcpyDeviceToDevice, stream);
    int total4 = N_EDGES * (dim >> 2);
    scatter_kernel<<<nb(total4), 256, 0, stream>>>(hw, m, src, dst,
                                                   total4, shift, (dim >> 2) - 1);
    gcn_finish_kernel<<<nb(N_NODES * dim), 256, 0, stream>>>(m, dinvI, bias, scaleNext,
                                                             oF, oB, N_NODES * dim,
                                                             shift, dim - 1, act);
  };

  // GCN layer 1: 256 -> 256, relu
  gemm(actA, Wg1, OBS, HID, 8, nullptr, hw, nullptr, 0);
  gcn_rest(HID, 8, bg1, dinvO, nullptr, actB, 1);
  // GCN layer 2: 256 -> 256, relu
  gemm(actB, Wg2, HID, HID, 8, nullptr, hw, nullptr, 0);
  gcn_rest(HID, 8, bg2, dinvO, nullptr, actA, 1);
  // GCN layer 3: 256 -> 128, sigmoid+1e-8; emits gnn_output (f32) + MLP input (bf16)
  gemm(actA, Wg3, HID, EMB, 7, nullptr, hw, nullptr, 0);
  gcn_rest(EMB, 7, bg3, nullptr, gnn, actB, 2);

  // MLP: 128 -> 512 -> 512 -> 64
  gemm(actB, W1, EMB,   MLP_H, 9, b1, nullptr, actA, 1);
  gemm(actA, W2, MLP_H, MLP_H, 9, b2, nullptr, actB, 1);
  gemm(actB, W3, MLP_H, ACT_D, 6, b3, hw, nullptr, 0);

  softmax64_kernel<<<(N_NODES + 7) / 8, 256, 0, stream>>>(hw, prob, N_NODES);
}
